// Mesh2_80985903334298
// MI455X (gfx1250) — matrix operand order repaired and checked by reading, not compiled, not run on hardware
//
#include <hip/hip_runtime.h>
#include <stdint.h>

typedef __attribute__((ext_vector_type(16))) _Float16 v16h;
typedef __attribute__((ext_vector_type(8)))  _Float16 v8h;
typedef __attribute__((ext_vector_type(4)))  _Float16 v4h;
typedef __attribute__((ext_vector_type(8)))  float    v8f;

#define C_DIM   256
#define O_DIM   512
#define BM      64        // nodes per block (4 m-tiles of 16)
#define NT      32        // n-tiles per GEMM (512/16)
#define KT_COMB 16        // K=512 in steps of 32
#define KT_AGG  8         // K=256 in steps of 32
#define THREADS 512       // 16 waves
#define ASTRIDE 72        // LDS row stride in halves (144B: 16B-aligned, conflict-free)

// B-fragment layout (halves): (((kt*NT + nt)*2 + plane)*32 + lane)*16
__device__ __forceinline__ const _Float16* fragPtr(const _Float16* base, int kt, int nt,
                                                   int plane, int lane) {
    return base + ((((kt * NT + nt) * 2 + plane) * 32 + lane) << 4);
}

// Split W[O][K] (f32, row-major) into hi/lo f16 fragments in this kernel's contiguous-k B-fragment order:
// lane l -> column o = nt*16 + (l&15), K run = kt*32 + (l&16 ? 16 : 0) .. +15
__global__ void prep_weights(const float* __restrict__ Wc, const float* __restrict__ Wa,
                             _Float16* __restrict__ fragC, _Float16* __restrict__ fragA) {
    int t = blockIdx.x * blockDim.x + threadIdx.x;
    const int nComb = KT_COMB * NT * 32;
    const float* W; _Float16* fb; int K; int rel;
    if (t < nComb) { W = Wc; fb = fragC; K = 512; rel = t; }
    else {
        rel = t - nComb;
        if (rel >= KT_AGG * NT * 32) return;
        W = Wa; fb = fragA; K = 256;
    }
    const int lane = rel & 31;
    const int nt   = (rel >> 5) & (NT - 1);
    const int kt   = rel >> 10;
    const int o    = nt * 16 + (lane & 15);
    const int kb   = kt * 32 + ((lane & 16) ? 16 : 0);
    const float* wr = W + (size_t)o * K + kb;
    _Float16* dh = fb + ((((kt * NT + nt) * 2 + 0) * 32 + lane) << 4);
    _Float16* dl = fb + ((((kt * NT + nt) * 2 + 1) * 32 + lane) << 4);
#pragma unroll
    for (int j = 0; j < 16; ++j) {
        float w = wr[j];
        _Float16 h = (_Float16)w;
        dh[j] = h;
        dl[j] = (_Float16)(w - (float)h);
    }
}

__global__ __launch_bounds__(THREADS, 1)
void mesh2_fused(const float* __restrict__ out1, const float* __restrict__ out2,
                 const int* __restrict__ nbr,
                 const float* __restrict__ bC, const float* __restrict__ bA,
                 const _Float16* __restrict__ fragC, const _Float16* __restrict__ fragA,
                 float* __restrict__ out3, float* __restrict__ out4, int N)
{
    __shared__ _Float16 sA[2][BM * ASTRIDE];   // [hi/lo][row*ASTRIDE + k]
    __shared__ int sIdx[BM][3];

    const int t    = threadIdx.x;
    const int lane = t & 31;
    const int wv   = t >> 5;          // wave 0..15
    const int wm   = wv >> 2;         // m-tile 0..3
    const int wg   = wv & 3;          // n-group 0..3 (8 n-tiles each)
    const int n0   = blockIdx.x * BM;

    // stage neighbor indices once (read only in phase B, after barriers)
    if (t < BM * 3) {
        int r = t / 3, j = t - r * 3;
        int n = n0 + r; if (n >= N) n = N - 1;
        sIdx[r][j] = nbr[(size_t)n * 3 + j];
    }

    // cooperative-staging coordinates: 64 rows x 32 cols, float4 per thread
    const int srow = t >> 3;
    const int scg  = (t & 7) << 2;
    int nrow = n0 + srow; if (nrow >= N) nrow = N - 1;

    // per-lane A-fragment base in LDS (contiguous k = 16*(lane/16) + s, matching B fragments)
    const int aoff = (lane & 16) ? 16 : 0;
    const _Float16* aBaseHi = &sA[0][(wm * 16 + (lane & 15)) * ASTRIDE + aoff];
    const _Float16* aBaseLo = &sA[1][(wm * 16 + (lane & 15)) * ASTRIDE + aoff];

    // ---------------- Phase A: out3 = [out1|out2] @ Wc^T + bC ----------------
    {
        v8f acc[8] = {};
        for (int ks = 0; ks < KT_COMB; ++ks) {
            const int kb = ks * 32;
            __syncthreads();
            {
                const float* src = (kb < C_DIM)
                    ? (out1 + (size_t)nrow * C_DIM + kb + scg)
                    : (out2 + (size_t)nrow * C_DIM + (kb - C_DIM) + scg);
                float4 v = *(const float4*)src;
                float vv[4] = { v.x, v.y, v.z, v.w };
                v4h h, l;
#pragma unroll
                for (int j = 0; j < 4; ++j) {
                    _Float16 hh = (_Float16)vv[j];
                    h[j] = hh;
                    l[j] = (_Float16)(vv[j] - (float)hh);
                }
                *(v4h*)&sA[0][srow * ASTRIDE + scg] = h;
                *(v4h*)&sA[1][srow * ASTRIDE + scg] = l;
            }
            __syncthreads();

            v8h c0 = *(const v8h*)(aBaseHi);
            v8h c1 = *(const v8h*)(aBaseHi + 8);
            v16h aHi = __builtin_shufflevector(c0, c1, 0,1,2,3,4,5,6,7,8,9,10,11,12,13,14,15);
            v8h d0 = *(const v8h*)(aBaseLo);
            v8h d1 = *(const v8h*)(aBaseLo + 8);
            v16h aLo = __builtin_shufflevector(d0, d1, 0,1,2,3,4,5,6,7,8,9,10,11,12,13,14,15);
#pragma unroll
            for (int i = 0; i < 8; ++i) {
                const int nt = wg * 8 + i;
                v16h bHi = *(const v16h*)fragPtr(fragC, ks, nt, 0, lane);
                v16h bLo = *(const v16h*)fragPtr(fragC, ks, nt, 1, lane);
                acc[i] = __builtin_amdgcn_wmma_f32_16x16x32_f16(false, aHi, false, bHi, (short)0, acc[i], false, false);
                acc[i] = __builtin_amdgcn_wmma_f32_16x16x32_f16(false, aLo, false, bHi, (short)0, acc[i], false, false);
                acc[i] = __builtin_amdgcn_wmma_f32_16x16x32_f16(false, aHi, false, bLo, (short)0, acc[i], false, false);
            }
        }
        // store out3 (+bias); C/D layout: VGPR r -> row r + (lane>=16 ? 8:0), col = lane&15
        const int rb = n0 + wm * 16 + ((lane & 16) ? 8 : 0);
#pragma unroll
        for (int i = 0; i < 8; ++i) {
            const int col = (wg * 8 + i) * 16 + (lane & 15);
            const float bias = bC[col];
#pragma unroll
            for (int r = 0; r < 8; ++r) {
                int g = rb + r;
                if (g < N) out3[(size_t)g * O_DIM + col] = acc[i][r] + bias;
            }
        }
    }

    // ---------------- Phase B: out4 = mean(out2 self+3 nbrs) @ Wa^T + bA ----------------
    {
        v8f acc[8] = {};
        for (int ks = 0; ks < KT_AGG; ++ks) {
            const int kb = ks * 32;
            __syncthreads();
            {
                const int i0 = sIdx[srow][0], i1 = sIdx[srow][1], i2 = sIdx[srow][2];
                float4 s0 = *(const float4*)(out2 + (size_t)nrow * C_DIM + kb + scg);
                float4 s1 = *(const float4*)(out2 + (size_t)i0   * C_DIM + kb + scg);
                float4 s2 = *(const float4*)(out2 + (size_t)i1   * C_DIM + kb + scg);
                float4 s3 = *(const float4*)(out2 + (size_t)i2   * C_DIM + kb + scg);
                float vv[4] = { 0.25f * (s0.x + s1.x + s2.x + s3.x),
                                0.25f * (s0.y + s1.y + s2.y + s3.y),
                                0.25f * (s0.z + s1.z + s2.z + s3.z),
                                0.25f * (s0.w + s1.w + s2.w + s3.w) };
                v4h h, l;
#pragma unroll
                for (int j = 0; j < 4; ++j) {
                    _Float16 hh = (_Float16)vv[j];
                    h[j] = hh;
                    l[j] = (_Float16)(vv[j] - (float)hh);
                }
                *(v4h*)&sA[0][srow * ASTRIDE + scg] = h;
                *(v4h*)&sA[1][srow * ASTRIDE + scg] = l;
            }
            __syncthreads();

            v8h c0 = *(const v8h*)(aBaseHi);
            v8h c1 = *(const v8h*)(aBaseHi + 8);
            v16h aHi = __builtin_shufflevector(c0, c1, 0,1,2,3,4,5,6,7,8,9,10,11,12,13,14,15);
            v8h d0 = *(const v8h*)(aBaseLo);
            v8h d1 = *(const v8h*)(aBaseLo + 8);
            v16h aLo = __builtin_shufflevector(d0, d1, 0,1,2,3,4,5,6,7,8,9,10,11,12,13,14,15);
#pragma unroll
            for (int i = 0; i < 8; ++i) {
                const int nt = wg * 8 + i;
                v16h bHi = *(const v16h*)fragPtr(fragA, ks, nt, 0, lane);
                v16h bLo = *(const v16h*)fragPtr(fragA, ks, nt, 1, lane);
                acc[i] = __builtin_amdgcn_wmma_f32_16x16x32_f16(false, aHi, false, bHi, (short)0, acc[i], false, false);
                acc[i] = __builtin_amdgcn_wmma_f32_16x16x32_f16(false, aLo, false, bHi, (short)0, acc[i], false, false);
                acc[i] = __builtin_amdgcn_wmma_f32_16x16x32_f16(false, aHi, false, bLo, (short)0, acc[i], false, false);
            }
        }
        const int rb = n0 + wm * 16 + ((lane & 16) ? 8 : 0);
#pragma unroll
        for (int i = 0; i < 8; ++i) {
            const int col = (wg * 8 + i) * 16 + (lane & 15);
            const float bias = bA[col];
#pragma unroll
            for (int r = 0; r < 8; ++r) {
                int g = rb + r;
                if (g < N) out4[(size_t)g * O_DIM + col] = acc[i][r] + bias;
            }
        }
    }
}

extern "C" void kernel_launch(void* const* d_in, const int* in_sizes, int n_in,
                              void* d_out, int out_size, void* d_ws, size_t ws_size,
                              hipStream_t stream) {
    const float* out1 = (const float*)d_in[0];
    const float* out2 = (const float*)d_in[1];
    const int*   nbr  = (const int*)d_in[2];
    const float* Wc   = (const float*)d_in[3];
    const float* bC   = (const float*)d_in[4];
    const float* Wa   = (const float*)d_in[5];
    const float* bA   = (const float*)d_in[6];
    const int N = in_sizes[0] / C_DIM;

    // workspace: pre-swizzled hi/lo f16 B-fragments (1 MB + 512 KB)
    _Float16* fragC = (_Float16*)d_ws;
    _Float16* fragA = fragC + (size_t)KT_COMB * NT * 2 * 32 * 16;

    const int prepThreads = (KT_COMB + KT_AGG) * NT * 32;   // 24576
    prep_weights<<<(prepThreads + 255) / 256, 256, 0, stream>>>(Wc, Wa, fragC, fragA);

    float* o3 = (float*)d_out;
    float* o4 = o3 + (size_t)N * O_DIM;
    const int blocks = (N + BM - 1) / BM;                   // 3125
    mesh2_fused<<<blocks, THREADS, 0, stream>>>(out1, out2, nbr, bC, bA,
                                                fragC, fragA, o3, o4, N);
}
